// W8A8Linear_69990787055930
// MI455X (gfx1250) — 8-bit matrix operand order repaired and checked by reading, compiled, not run on hardware
//
#include <hip/hip_runtime.h>
#include <hip/hip_fp16.h>

typedef __attribute__((ext_vector_type(8))) int       v8i;
typedef __attribute__((ext_vector_type(4))) int       v4i;
typedef __attribute__((ext_vector_type(2))) int       v2i;
typedef __attribute__((ext_vector_type(8))) _Float16  v8h;

#define MDIM 8192
#define KDIM 4096
#define NDIM 4096

__global__ __launch_bounds__(256) void quant_kernel(
    const _Float16* __restrict__ x,
    signed char*    __restrict__ xq,
    float*          __restrict__ xscale) {
  __shared__ float red[256];
  const int m = blockIdx.x;
  const int t = threadIdx.x;

  const _Float16* row = x + (size_t)m * KDIM + t * 16;
  v8h h0 = ((const v8h*)row)[0];
  v8h h1 = ((const v8h*)row)[1];

  float f[16];
#pragma unroll
  for (int i = 0; i < 8; ++i) { f[i] = (float)h0[i]; f[8 + i] = (float)h1[i]; }

  float amax = 0.0f;
#pragma unroll
  for (int i = 0; i < 16; ++i) amax = fmaxf(amax, fabsf(f[i]));

  red[t] = amax;
  __syncthreads();
#pragma unroll
  for (int s = 128; s > 0; s >>= 1) {
    if (t < s) red[t] = fmaxf(red[t], red[t + s]);
    __syncthreads();
  }
  const float absmax = red[0];
  const float inv    = 127.0f / fmaxf(absmax, 1e-20f);
  if (t == 0) xscale[m] = absmax * (1.0f / 127.0f);

  int pk[4];
#pragma unroll
  for (int g = 0; g < 4; ++g) {
    int v = 0;
#pragma unroll
    for (int j = 0; j < 4; ++j) {
      int q = (int)rintf(f[g * 4 + j] * inv);
      q = q > 127 ? 127 : (q < -128 ? -128 : q);
      v |= (q & 0xff) << (8 * j);
    }
    pk[g] = v;
  }
  v4i pv; pv[0] = pk[0]; pv[1] = pk[1]; pv[2] = pk[2]; pv[3] = pk[3];
  *(v4i*)(xq + (size_t)m * KDIM + t * 16) = pv;
}

__global__ __launch_bounds__(256, 2) void gemm_kernel(
    const signed char* __restrict__ xq,
    const signed char* __restrict__ w,
    const float*       __restrict__ xscale,
    const float*       __restrict__ wscale,
    const float*       __restrict__ mos,
    float*             __restrict__ out) {
  const int lane   = threadIdx.x & 31;
  const int wv     = threadIdx.x >> 5;
  const int waveM  = wv >> 1;
  const int waveN  = wv & 1;
  const int laneLo = lane & 15;
  const int laneHi = lane >> 4;
  const int M0 = blockIdx.y * 128 + waveM * 32;
  const int N0 = blockIdx.x * 128 + waveN * 64;

  v8i acc[2][4];
#pragma unroll
  for (int mi = 0; mi < 2; ++mi)
#pragma unroll
    for (int ni = 0; ni < 4; ++ni) acc[mi][ni] = (v8i){0, 0, 0, 0, 0, 0, 0, 0};

  const signed char* aptr0 = xq + (size_t)(M0 + laneLo) * KDIM + laneHi * 16;
  const signed char* aptr1 = aptr0 + (size_t)16 * KDIM;
  const signed char* bptr[4];
#pragma unroll
  for (int ni = 0; ni < 4; ++ni)
    bptr[ni] = w + (size_t)(N0 + ni * 16 + laneLo) * KDIM + laneHi * 16;

#pragma unroll 4
  for (int k0 = 0; k0 < KDIM; k0 += 64) {
    if ((k0 & 255) == 0) {
      __builtin_prefetch(aptr0 + k0 + 512, 0, 1);
      __builtin_prefetch(bptr[0] + k0 + 512, 0, 1);
    }

    v8i a[2];
    {
      const signed char* p0 = aptr0 + k0;
      const signed char* p1 = aptr1 + k0;
#pragma unroll
      for (int mi = 0; mi < 2; ++mi) {
        const signed char* p = mi ? p1 : p0;
        v2i t0 = *(const v2i*)(p);
        v2i t1 = *(const v2i*)(p + 8);
        v2i t2 = *(const v2i*)(p + 32);
        v2i t3 = *(const v2i*)(p + 40);
        v8i av;
        av[0] = t0[0]; av[1] = t0[1];
        av[2] = t1[0]; av[3] = t1[1];
        av[4] = t2[0]; av[5] = t2[1];
        av[6] = t3[0]; av[7] = t3[1];
        a[mi] = av;
      }
    }

    v8i b[4];
#pragma unroll
    for (int ni = 0; ni < 4; ++ni) {
      const signed char* p = bptr[ni] + k0;
      v4i t0 = *(const v4i*)(p);
      v4i t1 = *(const v4i*)(p + 32);
      v8i bv;
      bv[0] = t0[0]; bv[1] = t0[1]; bv[2] = t0[2]; bv[3] = t0[3];
      bv[4] = t1[0]; bv[5] = t1[1]; bv[6] = t1[2]; bv[7] = t1[3];
      b[ni] = bv;
    }

#pragma unroll
    for (int mi = 0; mi < 2; ++mi)
#pragma unroll
      for (int ni = 0; ni < 4; ++ni)
        acc[mi][ni] = __builtin_amdgcn_wmma_i32_16x16x64_iu8(
             true, a[mi],  true, b[ni],
            acc[mi][ni],  false,  false);
  }

  const float s = mos[0];
  float wsn[4];
#pragma unroll
  for (int ni = 0; ni < 4; ++ni) wsn[ni] = wscale[N0 + ni * 16 + laneLo];

#pragma unroll
  for (int mi = 0; mi < 2; ++mi) {
    const int rowBase = M0 + mi * 16 + laneHi * 8;
    float xsm[8];
#pragma unroll
    for (int r = 0; r < 8; ++r)
      xsm[r] = (float)(_Float16)xscale[rowBase + r];
#pragma unroll
    for (int ni = 0; ni < 4; ++ni) {
      const int col = N0 + ni * 16 + laneLo;
#pragma unroll
      for (int r = 0; r < 8; ++r) {
        float y = (float)acc[mi][ni][r] * s;
        _Float16 yh = (_Float16)y;
        out[(size_t)(rowBase + r) * NDIM + col] = (float)yh * wsn[ni] * xsm[r];
      }
    }
  }
}

extern "C" void kernel_launch(void* const* d_in, const int* in_sizes, int n_in,
                              void* d_out, int out_size, void* d_ws, size_t ws_size,
                              hipStream_t stream) {
  const _Float16*    x      = (const _Float16*)d_in[0];
  const signed char* w      = (const signed char*)d_in[1];
  const float*       wscale = (const float*)d_in[2];
  const float*       mos    = (const float*)d_in[3];
  float*             out    = (float*)d_out;

  signed char* xq = (signed char*)d_ws;
  float*       xs = (float*)((char*)d_ws + (size_t)MDIM * KDIM);

  quant_kernel<<<MDIM, 256, 0, stream>>>(x, xq, xs);

  dim3 grid(NDIM / 128, MDIM / 128);
  gemm_kernel<<<grid, 256, 0, stream>>>(xq, w, xs, wscale, mos, out);
}
